// RoutedEdgeClassifier_75617194213681
// MI455X (gfx1250) — matrix operand order repaired and checked by reading, not compiled, not run on hardware
//
#include <hip/hip_runtime.h>
#include <hip/hip_bf16.h>

// ---------------- types ----------------
typedef __attribute__((ext_vector_type(16))) _Float16 v16h;
typedef __attribute__((ext_vector_type(8)))  _Float16 v8h;
typedef __attribute__((ext_vector_type(8)))  float    v8f;

#define N_NODES 2048
#define N_EDGES 4096
#define D_NODE  128
#define D_EDGE  128
#define D_KV    384
#define N_HEADS 4
#define D_HEAD  32
#define N_KEEP  819    // int(0.4*2048)
#define E_KEEP  1638   // int(0.4*4096)

__device__ __forceinline__ v16h cat8(v8h lo, v8h hi) {
  v16h r;
#pragma unroll
  for (int i = 0; i < 8; ++i) { r[i] = lo[i]; r[i + 8] = hi[i]; }
  return r;
}
__device__ __forceinline__ v8h ld8h(const _Float16* p) { return *(const v8h*)p; }
__device__ __forceinline__ v8f wmma16(v16h a, v16h b, v8f c) {
  return __builtin_amdgcn_wmma_f32_16x16x32_f16(false, a, false, b, (short)0, c, false, false);
}

// ---------------- 1) importance scores: s = X @ w ----------------
__global__ void k_score(const float* __restrict__ X, const float* __restrict__ w,
                        float* __restrict__ s, int M, int K) {
  int i = blockIdx.x * blockDim.x + threadIdx.x;
  if (i >= M) return;
  float acc = 0.f;
  const float* row = X + (size_t)i * K;
#pragma unroll 8
  for (int k = 0; k < K; ++k) acc += row[k] * w[k];
  s[i] = acc;
}

// ---------------- 2) top-k mask by stable rank count ----------------
__global__ void k_topk_mask(const float* __restrict__ s, int* __restrict__ mask,
                            int n, int kkeep) {
  int i = blockIdx.x * blockDim.x + threadIdx.x;
  if (i >= n) return;
  float si = s[i];
  int cnt = 0;
#pragma unroll 4
  for (int j = 0; j < n; ++j) {
    float sj = s[j];
    cnt += (sj > si) || (sj == si && j < i);
  }
  mask[i] = (cnt < kkeep) ? 1 : 0;
}

// ---------------- 3) build wef (f32 + f16) and kv_in (f16) ----------------
__global__ void k_build(const float* __restrict__ ef, const float* __restrict__ nf,
                        const int* __restrict__ src, const int* __restrict__ dst,
                        const int* __restrict__ emask, const int* __restrict__ nmask,
                        float* __restrict__ wef32, _Float16* __restrict__ wef16,
                        _Float16* __restrict__ kvin16) {
  int e = blockIdx.x;
  int t = threadIdx.x;                       // 0..127
  int se = src[e], de = dst[e];
  int keep = emask[e] && nmask[se] && nmask[de];
  float w = keep ? ef[(size_t)e * D_EDGE + t] : 0.f;
  wef32[(size_t)e * D_EDGE + t] = w;
  _Float16 wh = (_Float16)w;
  wef16[(size_t)e * D_EDGE + t] = wh;
  _Float16* kv = kvin16 + (size_t)e * D_KV;
  kv[t]            = wh;
  kv[D_EDGE + t]   = (_Float16)nf[(size_t)se * D_NODE + t];
  kv[2*D_EDGE + t] = (_Float16)nf[(size_t)de * D_NODE + t];
}

// ---------------- 4) pack f32 weight [K,N] into WMMA B lane layout ----------------
// Per (nt, kk) tile: 32 lanes x 16 halfs. lane<16: col n=nt*16+lane, K rows kk*32+0..15
//                                         lane>=16: col n=nt*16+lane-16, K rows kk*32+16..31
__global__ void k_packB(const float* __restrict__ W, _Float16* __restrict__ Wp,
                        int K, int N) {
  int t = blockIdx.x * blockDim.x + threadIdx.x;
  int ksteps = K >> 5, ntiles = N >> 4;
  if (t >= ntiles * ksteps * 32) return;
  int lane = t & 31;
  int kk   = (t >> 5) % ksteps;
  int nt   = (t >> 5) / ksteps;
  int n  = nt * 16 + (lane & 15);
  int kb = kk * 32 + ((lane >> 4) << 4);
  _Float16* o = Wp + (size_t)((nt * ksteps + kk) * 32 + lane) * 16;
#pragma unroll
  for (int i = 0; i < 16; ++i) o[i] = (_Float16)W[(size_t)(kb + i) * N + n];
}

// ---------------- 5) WMMA GEMM: one wave computes a 16x64 strip -------------
// Fully unrolled K (template), one-step lookahead so next k-step's b128 loads
// overlap the current WMMAs. A tile reused across 4 accumulators.
template <int KSTEPS, bool BIAS, bool RES, bool GELU, bool WT>
__global__ void k_gemm(const _Float16* __restrict__ A, const _Float16* __restrict__ Bp,
                       const float* __restrict__ bias, const float* __restrict__ res,
                       _Float16* __restrict__ C, int M, int N) {
  const int lane = threadIdx.x & 31;
  const int wave = threadIdx.x >> 5;
  const int strip = blockIdx.x * (blockDim.x >> 5) + wave;
  const int nstrips = N >> 6;                       // 64-wide strips
  if (strip >= (M >> 4) * nstrips) return;
  const int mt = strip / nstrips, nq = strip % nstrips;
  const int hi = lane >> 4;
  const int ln = lane & 15;
  const int K = KSTEPS * 32;

  const _Float16* arow  = A + (size_t)(mt * 16 + ln) * K + hi * 16;
  const _Float16* bbase = Bp + (size_t)(nq * 4 * KSTEPS) * 512 + lane * 16;

  auto loadA = [&](int kk) {
    return cat8(ld8h(arow + kk * 32), ld8h(arow + kk * 32 + 8));
  };
  auto loadB = [&](int nt, int kk) {
    const _Float16* p = bbase + (size_t)(nt * KSTEPS + kk) * 512;
    return cat8(ld8h(p), ld8h(p + 8));
  };

  v8f acc[4] = {};
  v16h av = loadA(0);
  v16h bcur[4];
#pragma unroll
  for (int nt = 0; nt < 4; ++nt) bcur[nt] = loadB(nt, 0);

#pragma unroll
  for (int kk = 0; kk < KSTEPS; ++kk) {
    v16h avn;
    v16h bnxt[4];
    if (kk + 1 < KSTEPS) {                 // issue next tiles before waiting
      avn = loadA(kk + 1);
#pragma unroll
      for (int nt = 0; nt < 4; ++nt) bnxt[nt] = loadB(nt, kk + 1);
    }
#pragma unroll
    for (int nt = 0; nt < 4; ++nt) acc[nt] = wmma16(av, bcur[nt], acc[nt]);
    if (kk + 1 < KSTEPS) {
      av = avn;
#pragma unroll
      for (int nt = 0; nt < 4; ++nt) bcur[nt] = bnxt[nt];
    }
  }

#pragma unroll
  for (int nt = 0; nt < 4; ++nt) {
    const int ncol = (nq * 4 + nt) * 16 + ln;
    const float bia = BIAS ? bias[ncol] : 0.f;
#pragma unroll
    for (int r = 0; r < 8; ++r) {
      int mrow = mt * 16 + r + hi * 8;
      float v = acc[nt][r] + bia;
      if (RES) v += res[(size_t)mrow * N + ncol];
      if (GELU) {
        float x = v;
        v = 0.5f * x * (1.f + tanhf(0.7978845608028654f * (x + 0.044715f * x * x * x)));
      }
      if (WT) C[(size_t)ncol * M + mrow] = (_Float16)v;   // column-major (V^T)
      else    C[(size_t)mrow * N + ncol] = (_Float16)v;
    }
  }
}

// ---------------- 6) flash attention over adjacent edges ----------------
// grid.x = (E/16)*H, blockDim = 32 (one wave). Double-buffered K loads, V
// loads issued before the softmax pass so they hide under the VALU work.
// Softmax rows split across half-waves; stats combined with __shfl_xor.
__global__ void k_attn(const _Float16* __restrict__ Qh, const _Float16* __restrict__ Kh,
                       const _Float16* __restrict__ Vt,
                       const int* __restrict__ src, const int* __restrict__ dst,
                       _Float16* __restrict__ ctx, int E) {
  const int qt = blockIdx.x >> 2;
  const int h  = blockIdx.x & (N_HEADS - 1);
  const int lane = threadIdx.x;
  const int hi = lane >> 4, ln = lane & 15;
  const int qbase = qt * 16;

  __shared__ float Slds[16][32];
  __shared__ __align__(16) _Float16 Plds[16][32];
  __shared__ int ks_s[32], kd_s[32];

  // this lane helps softmax row ln; both half-waves hold its endpoints
  const int qsv = src[qbase + ln];
  const int qdv = dst[qbase + ln];

  // Q tile in A layout: row = ln, K chunk {hi*16..hi*16+15}
  const _Float16* qrow = Qh + (size_t)(qbase + ln) * D_EDGE + h * D_HEAD + hi * 16;
  v16h qa = cat8(ld8h(qrow), ld8h(qrow + 8));

  auto loadK = [&](int kb2, int sub) {     // 16 keys starting at kb2+sub
    const _Float16* p = Kh + (size_t)(kb2 + sub + ln) * D_EDGE + h * D_HEAD + hi * 16;
    return cat8(ld8h(p), ld8h(p + 8));
  };
  auto loadV = [&](int kb2, int sub) {     // dh columns sub..sub+15, keys kb2..+31
    const _Float16* p = Vt + (size_t)(h * D_HEAD + sub + ln) * E + kb2 + hi * 16;
    return cat8(ld8h(p), ld8h(p + 8));
  };

  // preload first K block + its endpoints
  v16h kt0 = loadK(0, 0), kt1 = loadK(0, 16);
  int ksv_c = src[lane], kdv_c = dst[lane];

  v8f O0 = {}, O1 = {};
  float mrun = -1e30f, lrun = 0.f;
  const float scale = 0.17677669529663687f;   // 1/sqrt(32)

  for (int kb = 0; kb < E; kb += 32) {
    ks_s[lane] = ksv_c;
    kd_s[lane] = kdv_c;

    // current block's V tiles: issue now, consumed after the softmax pass
    v16h vb0 = loadV(kb, 0), vb1 = loadV(kb, 16);

    // scores for two 16-key subtiles
    v8f S0 = {}, S1 = {};
    S0 = wmma16(qa, kt0, S0);
    S1 = wmma16(qa, kt1, S1);

    // next block's K tiles + endpoints: in flight during the softmax pass
    if (kb + 32 < E) {
      kt0 = loadK(kb + 32, 0);
      kt1 = loadK(kb + 32, 16);
      ksv_c = src[kb + 32 + lane];
      kdv_c = dst[kb + 32 + lane];
      __builtin_prefetch(Kh + (size_t)(kb + 64 + ln) * D_EDGE + h * D_HEAD, 0, 1);
    }

    // scatter C-layout scores to LDS rows
#pragma unroll
    for (int r = 0; r < 8; ++r) {
      Slds[r + hi * 8][ln]      = S0[r];
      Slds[r + hi * 8][16 + ln] = S1[r];
    }

    // parallel online softmax: lane handles row ln, columns hi*16..hi*16+15
    float sv[16];
    float lmax = -1e30f;
#pragma unroll
    for (int j = 0; j < 16; ++j) {
      int jj = hi * 16 + j;
      int ksv = ks_s[jj], kdv = kd_s[jj];
      bool adj = (qsv == ksv) | (qsv == kdv) | (qdv == ksv) | (qdv == kdv);
      float sc = adj ? Slds[ln][jj] * scale : -1e9f;
      sv[j] = sc;
      lmax = fmaxf(lmax, sc);
    }
    lmax = fmaxf(lmax, __shfl_xor(lmax, 16));
    float mnew = fmaxf(mrun, lmax);
    float corr = __expf(mrun - mnew);
    float lsum = 0.f;
#pragma unroll
    for (int j = 0; j < 16; ++j) {
      float p = __expf(sv[j] - mnew);
      lsum += p;
      Plds[ln][hi * 16 + j] = (_Float16)p;
    }
    lsum += __shfl_xor(lsum, 16);
    lrun = lrun * corr + lsum;
    mrun = mnew;

    // rescale accumulators; row r+hi*8's corr lives in lane r+hi*8
#pragma unroll
    for (int r = 0; r < 8; ++r) {
      float c = __shfl(corr, r + hi * 8);
      O0[r] *= c; O1[r] *= c;
    }

    // P in A layout straight from row-major LDS
    const _Float16* pr = &Plds[ln][hi * 16];
    v16h pa = cat8(ld8h(pr), ld8h(pr + 8));

    O0 = wmma16(pa, vb0, O0);
    O1 = wmma16(pa, vb1, O1);
  }

  float linv = 1.f / fmaxf(lrun, 1e-20f);
#pragma unroll
  for (int r = 0; r < 8; ++r) {
    float li = __shfl(linv, r + hi * 8);
    size_t row = (size_t)(qbase + r + hi * 8);
    ctx[row * D_EDGE + h * D_HEAD + ln]      = (_Float16)(O0[r] * li);
    ctx[row * D_EDGE + h * D_HEAD + 16 + ln] = (_Float16)(O1[r] * li);
  }
}

// ---------------- 7) classifier head: out[E,4] = h @ W2 + b2 ----------------
__global__ void k_head(const _Float16* __restrict__ Hf, const float* __restrict__ W2,
                       const float* __restrict__ b2, float* __restrict__ out, int E) {
  int t = blockIdx.x * blockDim.x + threadIdx.x;
  if (t >= E * 4) return;
  int e = t >> 2, c = t & 3;
  float acc = b2[c];
  const _Float16* row = Hf + (size_t)e * D_EDGE;
#pragma unroll 8
  for (int k = 0; k < D_EDGE; ++k) acc += (float)row[k] * W2[k * 4 + c];
  out[t] = acc;
}

// ---------------- host ----------------
extern "C" void kernel_launch(void* const* d_in, const int* in_sizes, int n_in,
                              void* d_out, int out_size, void* d_ws, size_t ws_size,
                              hipStream_t stream) {
  (void)in_sizes; (void)n_in; (void)out_size; (void)ws_size;
  const float* nf     = (const float*)d_in[0];
  const float* ef     = (const float*)d_in[1];
  const int*   eidx   = (const int*)d_in[2];
  const float* w_node = (const float*)d_in[3];
  const float* w_edge = (const float*)d_in[4];
  const float* Wq = (const float*)d_in[5];  const float* bq = (const float*)d_in[6];
  const float* Wk = (const float*)d_in[7];  const float* bk = (const float*)d_in[8];
  const float* Wv = (const float*)d_in[9];  const float* bv = (const float*)d_in[10];
  const float* Wo = (const float*)d_in[11]; const float* bo = (const float*)d_in[12];
  const float* W1 = (const float*)d_in[13]; const float* b1 = (const float*)d_in[14];
  const float* W2 = (const float*)d_in[15]; const float* b2 = (const float*)d_in[16];
  float* out = (float*)d_out;

  const int* src = eidx;
  const int* dst = eidx + N_EDGES;

  char* ws = (char*)d_ws;
  size_t cur = 0;
  auto alloc = [&](size_t bytes) -> void* {
    void* p = ws + cur;
    cur = (cur + bytes + 255) & ~(size_t)255;
    return p;
  };

  float*     ns     = (float*)alloc(N_NODES * 4);
  float*     es     = (float*)alloc(N_EDGES * 4);
  int*       nmask  = (int*)alloc(N_NODES * 4);
  int*       emask  = (int*)alloc(N_EDGES * 4);
  float*     wef32  = (float*)alloc((size_t)N_EDGES * D_EDGE * 4);
  _Float16*  wef16  = (_Float16*)alloc((size_t)N_EDGES * D_EDGE * 2);
  _Float16*  kvin16 = (_Float16*)alloc((size_t)N_EDGES * D_KV * 2);
  _Float16*  wq_p   = (_Float16*)alloc((size_t)(128/16)*(128/32)*512 * 2);
  _Float16*  wk_p   = (_Float16*)alloc((size_t)(128/16)*(384/32)*512 * 2);
  _Float16*  wv_p   = (_Float16*)alloc((size_t)(128/16)*(384/32)*512 * 2);
  _Float16*  wo_p   = (_Float16*)alloc((size_t)(128/16)*(128/32)*512 * 2);
  _Float16*  w1_p   = (_Float16*)alloc((size_t)(128/16)*(128/32)*512 * 2);
  _Float16*  Qh     = (_Float16*)alloc((size_t)N_EDGES * D_EDGE * 2);
  _Float16*  Khm    = (_Float16*)alloc((size_t)N_EDGES * D_EDGE * 2);
  _Float16*  Vt     = (_Float16*)alloc((size_t)N_EDGES * D_EDGE * 2);
  _Float16*  ctx16  = (_Float16*)alloc((size_t)N_EDGES * D_EDGE * 2);
  _Float16*  upd16  = (_Float16*)alloc((size_t)N_EDGES * D_EDGE * 2);
  _Float16*  h16    = (_Float16*)alloc((size_t)N_EDGES * D_EDGE * 2);

  // 1) scores
  k_score<<<(N_NODES + 255) / 256, 256, 0, stream>>>(nf, w_node, ns, N_NODES, D_NODE);
  k_score<<<(N_EDGES + 255) / 256, 256, 0, stream>>>(ef, w_edge, es, N_EDGES, D_EDGE);

  // 2) top-k masks
  k_topk_mask<<<(N_NODES + 255) / 256, 256, 0, stream>>>(ns, nmask, N_NODES, N_KEEP);
  k_topk_mask<<<(N_EDGES + 255) / 256, 256, 0, stream>>>(es, emask, N_EDGES, E_KEEP);

  // 3) wef / kv_in
  k_build<<<N_EDGES, 128, 0, stream>>>(ef, nf, src, dst, emask, nmask, wef32, wef16, kvin16);

  // 4) pack weights to WMMA B layout
  auto packB = [&](const float* W, _Float16* Wp, int K, int N) {
    int total = (N / 16) * (K / 32) * 32;
    k_packB<<<(total + 255) / 256, 256, 0, stream>>>(W, Wp, K, N);
  };
  packB(Wq, wq_p, 128, 128);
  packB(Wk, wk_p, 384, 128);
  packB(Wv, wv_p, 384, 128);
  packB(Wo, wo_p, 128, 128);
  packB(W1, w1_p, 128, 128);

  // 5) projections: one wave per 16x64 strip, 4 waves/block
  const int strips = (N_EDGES / 16) * (D_EDGE / 64);   // 512
  const int gblk   = strips / 4;                       // 128 blocks of 128 threads
  k_gemm<4,true,false,false,false><<<gblk,128,0,stream>>>(
      wef16, wq_p, bq, nullptr, Qh, N_EDGES, 128);
  k_gemm<12,true,false,false,false><<<gblk,128,0,stream>>>(
      kvin16, wk_p, bk, nullptr, Khm, N_EDGES, 128);
  k_gemm<12,true,false,false,true><<<gblk,128,0,stream>>>(    // V transposed
      kvin16, wv_p, bv, nullptr, Vt, N_EDGES, 128);

  // 6) attention
  k_attn<<<(N_EDGES / 16) * N_HEADS, 32, 0, stream>>>(Qh, Khm, Vt, src, dst, ctx16, N_EDGES);

  // 7) output projection + residual, then MLP with GELU
  k_gemm<4,true,true,false,false><<<gblk,128,0,stream>>>(
      ctx16, wo_p, bo, wef32, upd16, N_EDGES, 128);
  k_gemm<4,true,false,true,false><<<gblk,128,0,stream>>>(
      upd16, w1_p, b1, nullptr, h16, N_EDGES, 128);

  // 8) classifier head
  k_head<<<(N_EDGES * 4 + 255) / 256, 256, 0, stream>>>(h16, W2, b2, out, N_EDGES);
}
